// EchoStateNetwork_1649267441976
// MI455X (gfx1250) — matrix operand order repaired and checked by reading, not compiled, not run on hardware
//
#include <hip/hip_runtime.h>

// ---------------------------------------------------------------------------
// Echo State Network on MI455X (gfx1250, wave32, WMMA).
// Phase 1 (memory-bound, 262 MB stream ~11us @23.3TB/s):
//   P[t][b][n64] = tanh(inputs) @ W_in^T via bf16 WMMA, software-pipelined:
//   B-frags loaded into distinct regs (one clause, hidden behind tanh VALU),
//   next k-tile's A row prefetched before the WMMA chain.
// Phase 2 (latency-bound): 512 sequential state updates in one workgroup,
//   bf16 state double-buffered in LDS (1 barrier/step), W_res in registers,
//   P tile prefetched one step ahead.
// Phase 3: readout fused into phase-2 kernel.
// ---------------------------------------------------------------------------

typedef __attribute__((ext_vector_type(16))) __bf16        v16bf;
typedef __attribute__((ext_vector_type(8)))  float         v8f;
typedef __attribute__((ext_vector_type(4)))  float         f32x4;
typedef __attribute__((ext_vector_type(4)))  unsigned int  u32x4;

#define N_IN   2000
#define N_RES  50
#define N_OUT  500
#define BB     64
#define TT     512
#define KT_IN  63          // ceil(2000/32) k-tiles for the input GEMM

// workspace byte offsets
#define OFF_FWIN   0                         // 252 frags * 1024 B = 258048
#define OFF_FWRES  258048                    // 8 frags
#define OFF_FWOUT  266240                    // 64 frags
#define OFF_P      331776                    // 512*64*64 f32 = 8 MB

#define WMMA_BF16(A, B, C) \
  __builtin_amdgcn_wmma_f32_16x16x32_bf16(false, (A), false, (B), (short)0, (C), false, false)

union Frag { v16bf v; u32x4 q[2]; unsigned short u[16]; };

__device__ __forceinline__ unsigned short f2bf(float f) {
  unsigned u = __builtin_bit_cast(unsigned, f);
  unsigned r = u + 0x7FFFu + ((u >> 16) & 1u);   // round-to-nearest-even
  return (unsigned short)(r >> 16);
}

__device__ __forceinline__ float fast_tanh(float x) {
  // tanh(x) = (e^{2x}-1)/(e^{2x}+1) with hardware v_exp_f32 / v_rcp_f32
  x = fminf(fmaxf(x, -32.0f), 32.0f);
  float e = __builtin_amdgcn_exp2f(x * 2.8853900817779268f);  // 2*log2(e)
  return (e - 1.0f) * __builtin_amdgcn_rcpf(e + 1.0f);
}

// ---------------------------------------------------------------------------
// Prep: pre-swizzle W_in, W_res^T, W_out into bf16 WMMA B-fragments.
// B-fragment layout (16x16x32 bf16): lane L holds column n = nt*16 + (L&15);
// half h holds K = kt*32 + 16*(L>=16) + h. Out-of-range entries are 0, so
// all N/K padding contributes exact zeros downstream.
// ---------------------------------------------------------------------------
__global__ void esn_prep(const float* __restrict__ Win,
                         const float* __restrict__ Wres,
                         const float* __restrict__ Wout,
                         unsigned short* __restrict__ fWin,
                         unsigned short* __restrict__ fWres,
                         unsigned short* __restrict__ fWout) {
  int idx = blockIdx.x * blockDim.x + threadIdx.x;
  if (idx < KT_IN * 4 * 32) {                      // W_in: 63 kt x 4 nt
    int lane = idx & 31, f = idx >> 5;
    int nt = f & 3, kt = f >> 2;
    int n = nt * 16 + (lane & 15);
    int kbase = kt * 32 + ((lane >> 4) << 4);
    unsigned short* dst = fWin + (size_t)f * 512 + lane * 16;
#pragma unroll
    for (int h = 0; h < 16; ++h) {
      int k = kbase + h;                           // B[k][n] = W_in[n][k]
      dst[h] = f2bf((n < N_RES && k < N_IN) ? Win[n * N_IN + k] : 0.0f);
    }
    return;
  }
  idx -= KT_IN * 4 * 32;
  if (idx < 2 * 4 * 32) {                          // W_res: 2 kt x 4 nt
    int lane = idx & 31, f = idx >> 5;
    int nt = f & 3, kt = f >> 2;
    int n = nt * 16 + (lane & 15);
    int kbase = kt * 32 + ((lane >> 4) << 4);
    unsigned short* dst = fWres + (size_t)f * 512 + lane * 16;
#pragma unroll
    for (int h = 0; h < 16; ++h) {
      int k = kbase + h;                           // B[k][n] = W_res[n][k]
      dst[h] = f2bf((n < N_RES && k < N_RES) ? Wres[n * N_RES + k] : 0.0f);
    }
    return;
  }
  idx -= 2 * 4 * 32;
  if (idx < 2 * 32 * 32) {                         // W_out: 2 kt x 32 nt
    int lane = idx & 31, f = idx >> 5;
    int nt = f & 31, kt = f >> 5;
    int n = nt * 16 + (lane & 15);
    int kbase = kt * 32 + ((lane >> 4) << 4);
    unsigned short* dst = fWout + (size_t)f * 512 + lane * 16;
#pragma unroll
    for (int h = 0; h < 16; ++h) {
      int k = kbase + h;                           // B[k][n] = W_out[k][n]
      dst[h] = f2bf((n < N_OUT && k < N_RES) ? Wout[k * N_OUT + n] : 0.0f);
    }
    return;
  }
}

// ---------------------------------------------------------------------------
// Phase 1: P = tanh(inputs) @ W_in^T.  M = 32768 rows (b*512+t), N = 64(pad),
// K = 2000 (pad 2016).  One 16-row M-tile per wave, 4 f32 accumulator tiles.
// A-fragment layout (16-bit A 16x32): lane L holds row m=L%16; halves 0..15
// cover K = 16*(L>=16)+0..15 (matches B-fragments) — two
// contiguous 32-B runs -> 4x b128 loads per lane, coalesced per row.
// ---------------------------------------------------------------------------
__device__ __forceinline__ void load_arow(const float* __restrict__ arow,
                                          int k0, int hi, float va[16]) {
  int ka = k0 + hi * 16;                           // halves 0..7
  int kb = k0 + hi * 16 + 8;                       // halves 8..15
  if (ka < N_IN) {
    f32x4 x0 = *reinterpret_cast<const f32x4*>(arow + ka);
    f32x4 x1 = *reinterpret_cast<const f32x4*>(arow + ka + 4);
#pragma unroll
    for (int j = 0; j < 4; ++j) { va[j] = x0[j]; va[4 + j] = x1[j]; }
  } else {
#pragma unroll
    for (int j = 0; j < 8; ++j) va[j] = 0.0f;
  }
  if (kb < N_IN) {
    f32x4 y0 = *reinterpret_cast<const f32x4*>(arow + kb);
    f32x4 y1 = *reinterpret_cast<const f32x4*>(arow + kb + 4);
#pragma unroll
    for (int j = 0; j < 4; ++j) { va[8 + j] = y0[j]; va[12 + j] = y1[j]; }
  } else {
#pragma unroll
    for (int j = 0; j < 8; ++j) va[8 + j] = 0.0f;
  }
}

__global__ __launch_bounds__(256) void esn_proj(const float* __restrict__ inp,
                                                const unsigned short* __restrict__ fWin,
                                                float* __restrict__ P) {
  int lane = threadIdx.x & 31;
  int wave = threadIdx.x >> 5;
  int mtile = blockIdx.x * 8 + wave;               // 0..2047
  int mrow = lane & 15, hi = lane >> 4;
  const float* arow = inp + (size_t)(mtile * 16 + mrow) * N_IN;
  const u32x4* bq = reinterpret_cast<const u32x4*>(fWin);

  v8f acc0 = {}, acc1 = {}, acc2 = {}, acc3 = {};

  float va[16];
  load_arow(arow, 0, hi, va);                      // prologue: A for kt=0

  for (int kt = 0; kt < KT_IN; ++kt) {
    // 1) issue all 4 B-fragment loads into DISTINCT regs (one clause);
    //    their latency hides behind the tanh chain below.
    const u32x4* bb = bq + (size_t)(kt * 4) * 64 + lane * 2;
    Frag B0, B1, B2, B3;
    B0.q[0] = bb[0];   B0.q[1] = bb[1];
    B1.q[0] = bb[64];  B1.q[1] = bb[65];
    B2.q[0] = bb[128]; B2.q[1] = bb[129];
    B3.q[0] = bb[192]; B3.q[1] = bb[193];

    // 2) tanh + pack current A fragment (~16 exp/rcp chains of VALU work)
    Frag A;
#pragma unroll
    for (int h = 0; h < 16; ++h) A.u[h] = f2bf(fast_tanh(va[h]));

    // 3) prefetch next k-tile's A rows before the WMMA chain
    if (kt + 1 < KT_IN) load_arow(arow, (kt + 1) * 32, hi, va);

    // 4) accumulate
    acc0 = WMMA_BF16(A.v, B0.v, acc0);
    acc1 = WMMA_BF16(A.v, B1.v, acc1);
    acc2 = WMMA_BF16(A.v, B2.v, acc2);
    acc3 = WMMA_BF16(A.v, B3.v, acc3);
  }

  // store as P[t][b][n64] for contiguous access in the recurrence
#pragma unroll
  for (int r = 0; r < 8; ++r) {
    int mg = mtile * 16 + r + 8 * hi;              // global row = b*512 + t
    int t = mg & (TT - 1), b = mg >> 9;
    float* prow = P + (((size_t)t * BB + b) << 6);
    prow[0 * 16 + mrow] = acc0[r];
    prow[1 * 16 + mrow] = acc1[r];
    prow[2 * 16 + mrow] = acc2[r];
    prow[3 * 16 + mrow] = acc3[r];
  }
}

// ---------------------------------------------------------------------------
// Phase 2+3: sequential recurrence (single workgroup, 16 waves, one 16x16
// tile each; state [64][64] bf16 double-buffered in LDS -> 1 barrier/step;
// W_res fragments live in registers; P tile prefetched one step ahead).
// Then the readout GEMM (4x32 tiles over the 16 waves) fused at the end.
// ---------------------------------------------------------------------------
__global__ __launch_bounds__(512) void esn_recur(const float* __restrict__ P,
                                                 const unsigned short* __restrict__ fWres,
                                                 const unsigned short* __restrict__ fWout,
                                                 const float* __restrict__ bias,
                                                 float* __restrict__ out) {
  __shared__ unsigned short st[2][BB * 64];
  int tid = threadIdx.x;
  int lane = tid & 31, wave = tid >> 5;
  int mt = wave >> 2, nt = wave & 3;
  int mrow = lane & 15, hi = lane >> 4;

  // W_res B-fragments (kt = 0,1) held in registers for the whole loop
  const u32x4* rq = reinterpret_cast<const u32x4*>(fWres);
  Frag WB0, WB1;
  WB0.q[0] = rq[(0 * 4 + nt) * 64 + lane * 2];
  WB0.q[1] = rq[(0 * 4 + nt) * 64 + lane * 2 + 1];
  WB1.q[0] = rq[(1 * 4 + nt) * 64 + lane * 2];
  WB1.q[1] = rq[(1 * 4 + nt) * 64 + lane * 2 + 1];

  for (int i = tid; i < BB * 64; i += 512) st[0][i] = 0;  // state0 = 0
  __syncthreads();

  float pv[8];
  {
    const float* prow = P;                                 // t = 0
#pragma unroll
    for (int r = 0; r < 8; ++r)
      pv[r] = prow[(mt * 16 + r + 8 * hi) * 64 + nt * 16 + mrow];
  }

  for (int t = 0; t < TT; ++t) {
    float pn[8];                                           // prefetch t+1
    if (t + 1 < TT) {
      const float* prow = P + ((size_t)(t + 1) << 12);
#pragma unroll
      for (int r = 0; r < 8; ++r)
        pn[r] = prow[(mt * 16 + r + 8 * hi) * 64 + nt * 16 + mrow];
    } else {
#pragma unroll
      for (int r = 0; r < 8; ++r) pn[r] = 0.0f;
    }

    // A-fragments: state rows mt*16..mt*16+15, K runs of 8 halves (16 B each)
    const u32x4* s =
        reinterpret_cast<const u32x4*>(&st[t & 1][(mt * 16 + mrow) * 64]);
    Frag A0, A1;
    A0.q[0] = s[2 * hi];     A0.q[1] = s[2 * hi + 1];
    A1.q[0] = s[4 + 2 * hi]; A1.q[1] = s[5 + 2 * hi];

    v8f c = {};
    c = WMMA_BF16(A0.v, WB0.v, c);
    c = WMMA_BF16(A1.v, WB1.v, c);

    unsigned short* sw = &st[(t + 1) & 1][0];
#pragma unroll
    for (int r = 0; r < 8; ++r)
      sw[(mt * 16 + r + 8 * hi) * 64 + nt * 16 + mrow] =
          f2bf(fast_tanh(c[r] + pv[r]));
    __syncthreads();
#pragma unroll
    for (int r = 0; r < 8; ++r) pv[r] = pn[r];
  }

  // readout: final state is in st[0] (TT even); 128 tiles over 16 waves
  const u32x4* oq = reinterpret_cast<const u32x4*>(fWout);
  for (int j = 0; j < 8; ++j) {
    int tile = wave * 8 + j;
    int rmt = tile >> 5, rnt = tile & 31;
    const u32x4* s =
        reinterpret_cast<const u32x4*>(&st[0][(rmt * 16 + mrow) * 64]);
    Frag A0, A1, B0, B1;
    A0.q[0] = s[2 * hi];     A0.q[1] = s[2 * hi + 1];
    A1.q[0] = s[4 + 2 * hi]; A1.q[1] = s[5 + 2 * hi];
    B0.q[0] = oq[(0 * 32 + rnt) * 64 + lane * 2];
    B0.q[1] = oq[(0 * 32 + rnt) * 64 + lane * 2 + 1];
    B1.q[0] = oq[(1 * 32 + rnt) * 64 + lane * 2];
    B1.q[1] = oq[(1 * 32 + rnt) * 64 + lane * 2 + 1];
    v8f c = {};
    c = WMMA_BF16(A0.v, B0.v, c);
    c = WMMA_BF16(A1.v, B1.v, c);
    int n = rnt * 16 + mrow;
    if (n < N_OUT) {
      float bn = bias[n];
#pragma unroll
      for (int r = 0; r < 8; ++r) {
        int b = rmt * 16 + r + 8 * hi;
        out[b * N_OUT + n] = fast_tanh(c[r] + bn);
      }
    }
  }
}

extern "C" void kernel_launch(void* const* d_in, const int* in_sizes, int n_in,
                              void* d_out, int out_size, void* d_ws, size_t ws_size,
                              hipStream_t stream) {
  (void)in_sizes; (void)n_in; (void)out_size; (void)ws_size;
  const float* inputs  = (const float*)d_in[0];   // [64,512,2000]
  const float* W_in    = (const float*)d_in[1];   // [50,2000]
  const float* W_res   = (const float*)d_in[2];   // [50,50]
  const float* W_out_k = (const float*)d_in[3];   // [50,500]
  const float* W_out_b = (const float*)d_in[4];   // [500]
  float* out = (float*)d_out;                     // [64,500]

  char* ws = (char*)d_ws;                         // needs ~8.4 MB
  unsigned short* fWin  = (unsigned short*)(ws + OFF_FWIN);
  unsigned short* fWres = (unsigned short*)(ws + OFF_FWRES);
  unsigned short* fWout = (unsigned short*)(ws + OFF_FWOUT);
  float* P = (float*)(ws + OFF_P);

  esn_prep<<<41, 256, 0, stream>>>(W_in, W_res, W_out_k, fWin, fWres, fWout);
  esn_proj<<<256, 256, 0, stream>>>(inputs, fWin, P);
  esn_recur<<<1, 512, 0, stream>>>(P, fWres, fWout, W_out_b, out);
}
